// GATLayer_63857573757390
// MI455X (gfx1250) — compile-verified, matrix operand order checked by reading, not run on hardware
//
#include <hip/hip_runtime.h>
#include <math.h>

// ---------------------------------------------------------------------------
// GAT layer for MI455X (gfx1250, wave32, WMMA).
//   Wh = h @ W^T                        (f16 WMMA, f32 accumulate)
//   s1 = Wh@a1, s2 = Wh@a2, smax = max(s2)
//   out[i,:] = sum_j exp(lrelu(s1_i+s2_j) - lrelu(s1_i+smax)) * Wh[j,:] / Z_i
// The 8192x8192 attention matrix is rank-1 structured and never materialized.
// adj (d_in[1]) is unused by the reference math and is never read.
// ---------------------------------------------------------------------------

typedef __attribute__((ext_vector_type(16))) _Float16 v16h;
typedef __attribute__((ext_vector_type(8)))  float    v8f;
typedef __attribute__((ext_vector_type(2)))  _Float16 hpair;

#define N_NODES 8192
#define IN_F    512
#define OUT_F   64
#define ALPHA   0.2f

__device__ __forceinline__ float lrelu(float x) {
    // alpha < 1  =>  leaky_relu(x) == max(x, alpha*x)   (v_mul + v_max)
    return fmaxf(x, ALPHA * x);
}

// K offset for the 16-bit 16x32 A/B fragment layout (ISA 7.12.2):
// VGPR v (pair slots 2v,2v+1), lane half h: k0 = 2*(v&3) + 8*h + 16*(v>>2)
__device__ __forceinline__ int frag_k0(int v, int half) {
    return 2 * (v & 3) + 8 * half + 16 * (v >> 2);
}

// ---------------------------------------------------------------------------
// Kernel 1: Wh = h @ W^T  (8192x512 @ 512x64)
// One wave per 16x16 output tile: block b -> rows 16b, wave w -> feats 16w.
// Also emits WhT (f16, feature-major [64][8192]) so the attention kernel's
// B fragments are contiguous dword loads (and the epilogue's 8 consecutive
// f16 stores merge into one global_store_b128).
// ---------------------------------------------------------------------------
__global__ __launch_bounds__(128) void wh_kernel(const float* __restrict__ h,
                                                 const float* __restrict__ W,
                                                 float*    __restrict__ Wh,
                                                 _Float16* __restrict__ WhT) {
    const int lane = threadIdx.x & 31;
    const int wave = threadIdx.x >> 5;
    const int half = lane >> 4;
    const int ln   = lane & 15;
    const int j0   = blockIdx.x * 16;   // node-row tile
    const int ft   = wave * 16;         // feature tile

    v8f acc = {};
    for (int kb = 0; kb < IN_F; kb += 32) {
        v16h A, B;
#pragma unroll
        for (int v = 0; v < 8; ++v) {
            const int k0 = frag_k0(v, half);
            // A: h[j0+ln][kb+k0 .. +1]   (lane ln holds row M=ln in both halves)
            const float2 ha = *(const float2*)(h + (size_t)(j0 + ln) * IN_F + kb + k0);
            A[2 * v]     = (_Float16)ha.x;
            A[2 * v + 1] = (_Float16)ha.y;
            // B[k][n] = W^T[k][n] = W[ft+ln][kb+k0 .. +1]
            const float2 wb = *(const float2*)(W + (size_t)(ft + ln) * IN_F + kb + k0);
            B[2 * v]     = (_Float16)wb.x;
            B[2 * v + 1] = (_Float16)wb.y;
        }
        acc = __builtin_amdgcn_wmma_f32_16x16x32_f16(false, A, false, B,
                                                     (short)0, acc, false, false);
    }
    // C layout: VGPR r, lane l -> (M = r + 8*(l>>4), N = l&15)
#pragma unroll
    for (int r = 0; r < 8; ++r) {
        const int m = r + 8 * half;
        const float val = acc[r];
        Wh [(size_t)(j0 + m) * OUT_F + ft + ln] = val;
        WhT[(size_t)(ft + ln) * N_NODES + j0 + m] = (_Float16)val;
    }
}

// ---------------------------------------------------------------------------
// Kernel 2: s1[j] = Wh[j,:] . a[0:64],  s2[j] = Wh[j,:] . a[64:128]
// ---------------------------------------------------------------------------
__global__ __launch_bounds__(256) void s_kernel(const float* __restrict__ Wh,
                                                const float* __restrict__ a,
                                                float* __restrict__ s1,
                                                float* __restrict__ s2) {
    const int j = blockIdx.x * blockDim.x + threadIdx.x;
    if (j >= N_NODES) return;
    const float* row = Wh + (size_t)j * OUT_F;
    float acc1 = 0.f, acc2 = 0.f;
#pragma unroll 8
    for (int f = 0; f < OUT_F; ++f) {
        const float w = row[f];
        acc1 += w * a[f];
        acc2 += w * a[OUT_F + f];
    }
    s1[j] = acc1;
    s2[j] = acc2;
}

// ---------------------------------------------------------------------------
// Kernel 3: smax = max_j s2[j]  (single block; lrelu monotonic =>
// row max of e is lrelu(s1[i] + smax) -- no online softmax needed)
// ---------------------------------------------------------------------------
__global__ __launch_bounds__(256) void smax_kernel(const float* __restrict__ s2,
                                                   float* __restrict__ smax) {
    __shared__ float red[256];
    float m = -3.402823466e38f;
    for (int j = threadIdx.x; j < N_NODES; j += 256) m = fmaxf(m, s2[j]);
    red[threadIdx.x] = m;
    __syncthreads();
    for (int s = 128; s > 0; s >>= 1) {
        if (threadIdx.x < (unsigned)s)
            red[threadIdx.x] = fmaxf(red[threadIdx.x], red[threadIdx.x + s]);
        __syncthreads();
    }
    if (threadIdx.x == 0) smax[0] = red[0];
}

// ---------------------------------------------------------------------------
// Kernel 4: out = softmax_rows(lrelu(s1[:,None]+s2[None,:])) @ Wh
// Wave: TWO 16-row tiles x 16 feats, so each B fragment (Wh tile) feeds two
// WMMAs. j-loop unrolled x2 => 4 v_wmma per loop body; two independent
// accumulate chains let exp/cvt co-execute with WMMA issue.
//   A fragments = exp(lrelu(s1_i+s2_j) - m_i) computed in-register (f16)
//   B fragment  = WhT dword pairs (contiguous: feature-major storage)
// Denominators folded into the same pass (lane l & l+16 hold the same row ->
// one shfl_xor(16) completes each row sum).
// ---------------------------------------------------------------------------
__global__ __launch_bounds__(128) void attn_kernel(const _Float16* __restrict__ WhT,
                                                   const float* __restrict__ s1,
                                                   const float* __restrict__ s2,
                                                   const float* __restrict__ smaxp,
                                                   float* __restrict__ out) {
    const int lane = threadIdx.x & 31;
    const int wave = threadIdx.x >> 5;
    const int half = lane >> 4;
    const int ln   = lane & 15;
    const int i0   = blockIdx.x * 32;   // two 16-row output tiles
    const int ft   = wave * 16;         // feature tile

    const float smax = smaxp[0];
    const float s1a = s1[i0 + ln];            // row tile 0: lane's A-row
    const float s1b = s1[i0 + 16 + ln];       // row tile 1
    const float ma  = lrelu(s1a + smax);      // exact row maxes of e
    const float mb  = lrelu(s1b + smax);

    const _Float16* bbase = WhT + (size_t)(ft + ln) * N_NODES;

    v8f   accA = {}, accB = {};
    float rpA = 0.0f, rpB = 0.0f;

#pragma unroll 2
    for (int j0 = 0; j0 < N_NODES; j0 += 32) {
        v16h A0, A1, B;
#pragma unroll
        for (int v = 0; v < 8; ++v) {
            const int k0 = frag_k0(v, half);
            const float2 sv = *(const float2*)(s2 + j0 + k0);
            const float la0 = lrelu(s1a + sv.x), la1 = lrelu(s1a + sv.y);
            const float lb0 = lrelu(s1b + sv.x), lb1 = lrelu(s1b + sv.y);
            const float pa0 = __expf(la0 - ma), pa1 = __expf(la1 - ma);
            const float pb0 = __expf(lb0 - mb), pb1 = __expf(lb1 - mb);
            rpA += pa0 + pa1;
            rpB += pb0 + pb1;
            A0[2 * v]     = (_Float16)pa0;
            A0[2 * v + 1] = (_Float16)pa1;
            A1[2 * v]     = (_Float16)pb0;
            A1[2 * v + 1] = (_Float16)pb1;
            const hpair wb = *(const hpair*)(bbase + j0 + k0);
            B[2 * v]     = wb[0];
            B[2 * v + 1] = wb[1];
        }
        if (j0 + 32 < N_NODES)
            __builtin_prefetch(bbase + j0 + 32, 0, 1);   // global_prefetch_b8
        accA = __builtin_amdgcn_wmma_f32_16x16x32_f16(false, A0, false, B,
                                                      (short)0, accA, false, false);
        accB = __builtin_amdgcn_wmma_f32_16x16x32_f16(false, A1, false, B,
                                                      (short)0, accB, false, false);
    }

    // Row sums: lane l holds the partial for row (l&15); the other half-wave
    // holds the complementary K range of the same row.
    const float rsA = rpA + __shfl_xor(rpA, 16, 32);
    const float rsB = rpB + __shfl_xor(rpB, 16, 32);
#pragma unroll
    for (int r = 0; r < 8; ++r) {
        const int m = r + 8 * half;
        const float dA = __shfl(rsA, m, 32);   // lane m holds rowsum[m]
        const float dB = __shfl(rsB, m, 32);
        out[(size_t)(i0 + m) * OUT_F + ft + ln]      = accA[r] / dA;
        out[(size_t)(i0 + 16 + m) * OUT_F + ft + ln] = accB[r] / dB;
    }
}

// ---------------------------------------------------------------------------
extern "C" void kernel_launch(void* const* d_in, const int* in_sizes, int n_in,
                              void* d_out, int out_size, void* d_ws, size_t ws_size,
                              hipStream_t stream) {
    const float* h = (const float*)d_in[0];
    // d_in[1] = adj : unused by the reference computation
    const float* W = (const float*)d_in[2];
    const float* a = (const float*)d_in[3];
    float* out = (float*)d_out;

    char* ws = (char*)d_ws;
    float*    Wh   = (float*)ws;      ws += (size_t)N_NODES * OUT_F * sizeof(float);
    _Float16* WhT  = (_Float16*)ws;   ws += (size_t)OUT_F * N_NODES * sizeof(_Float16);
    float*    s1   = (float*)ws;      ws += (size_t)N_NODES * sizeof(float);
    float*    s2   = (float*)ws;      ws += (size_t)N_NODES * sizeof(float);
    float*    smax = (float*)ws;
    (void)ws_size; (void)in_sizes; (void)n_in; (void)out_size;

    wh_kernel  <<<N_NODES / 16, 128, 0, stream>>>(h, W, Wh, WhT);
    s_kernel   <<<N_NODES / 256, 256, 0, stream>>>(Wh, a, s1, s2);
    smax_kernel<<<1, 256, 0, stream>>>(s2, smax);
    attn_kernel<<<N_NODES / 32, 128, 0, stream>>>(WhT, s1, s2, smax, out);
}
